// SelfAttention_51505247814196
// MI455X (gfx1250) — matrix operand order repaired and checked by reading, not compiled, not run on hardware
//
#include <hip/hip_runtime.h>

typedef _Float16 f16;
typedef _Float16 v16h __attribute__((ext_vector_type(16)));
typedef _Float16 v8h  __attribute__((ext_vector_type(8)));
typedef float    v8f  __attribute__((ext_vector_type(8)));

#define WMMA16(a,b,c) __builtin_amdgcn_wmma_f32_16x16x32_f16(false,(a),false,(b),(short)0,(c),false,false)

static constexpr int Bd = 4, T = 1024, C = 1024, H = 16, HD = 64;

__device__ inline v8f vzero8() {
  v8f z;
#pragma unroll
  for (int i = 0; i < 8; ++i) z[i] = 0.f;
  return z;
}

// A-matrix fragment (16x32 f16): lane holds row (lane&15); halves are
// K = (lane>=16)*16 + {0..15}, i.e. 16 contiguous halves.
__device__ inline v16h make_afrag(const f16* __restrict__ p) {
  v8h lo = *(const v8h*)p;
  v8h hi = *(const v8h*)(p + 8);
  v16h r;
#pragma unroll
  for (int i = 0; i < 8; ++i) { r[i] = lo[i]; r[i + 8] = hi[i]; }
  return r;
}

// B-matrix fragment (32x16 f16): lane holds column (lane&15); halves are
// K = (lane>=16)*16 + {0..15}, i.e. 16 contiguous halves (32 bytes).
__device__ inline v16h make_bfrag(const f16* __restrict__ p) {
  v8h lo = *(const v8h*)p;
  v8h hi = *(const v8h*)(p + 8);
  v16h r;
#pragma unroll
  for (int i = 0; i < 8; ++i) { r[i] = lo[i]; r[i + 8] = hi[i]; }
  return r;
}

// ---------------- conversion kernels ----------------
__global__ void cvt_f32_f16(const float* __restrict__ s, f16* __restrict__ d, int n) {
  int i = blockIdx.x * 256 + threadIdx.x;
  if (i < n) d[i] = (f16)s[i];
}

// dst[c*rows + r] = src[r*cols + c]
__global__ void cvt_t_f16(const float* __restrict__ s, f16* __restrict__ d, int rows, int cols) {
  int i = blockIdx.x * 256 + threadIdx.x;
  if (i < rows * cols) {
    int r = i / cols, c = i % cols;
    d[(size_t)c * rows + r] = (f16)s[i];
  }
}

// ---------------- QKV GEMM:  qkv = xh @ WaT^T + b ----------------
// one wave: 32x64 output tile; per K=32 step: 12 grouped B128 loads then 8 WMMAs.
// No cross-iteration fragment state -> no register copies, good occupancy.
__global__ __launch_bounds__(256) void qkv_gemm(
    const f16* __restrict__ xh, const f16* __restrict__ WaT,
    const float* __restrict__ bias,
    f16* __restrict__ qh, f16* __restrict__ kh, f16* __restrict__ vT) {
  const int lane = threadIdx.x & 31, wid = threadIdx.x >> 5;
  const int w = blockIdx.x * 8 + wid;          // 6144 waves
  const int mt = w / 48, nt = w % 48;
  const int m0 = mt * 32, n0 = nt * 64;
  const int lr = lane & 15, lh = lane >> 4;

  v8f acc[2][4];
#pragma unroll
  for (int mi = 0; mi < 2; ++mi)
#pragma unroll
    for (int s = 0; s < 4; ++s) acc[mi][s] = vzero8();

  const f16* arow0 = xh + (size_t)(m0 + lr) * C + lh * 16;
  const f16* arow1 = xh + (size_t)(m0 + 16 + lr) * C + lh * 16;
  const f16* brow[4];
#pragma unroll
  for (int s = 0; s < 4; ++s) brow[s] = WaT + (size_t)(n0 + s * 16 + lr) * C + lh * 16;

  for (int k0 = 0; k0 < C; k0 += 32) {
    __builtin_prefetch(arow0 + k0 + 32, 0, 3);     // global_prefetch_b8, one step ahead
    __builtin_prefetch(brow[0] + k0 + 32, 0, 3);
    // all fragment loads for this K-step issued together
    v16h a0 = make_afrag(arow0 + k0);
    v16h a1 = make_afrag(arow1 + k0);
    v16h b[4];
#pragma unroll
    for (int s = 0; s < 4; ++s) b[s] = make_bfrag(brow[s] + k0);
#pragma unroll
    for (int s = 0; s < 4; ++s) {
      acc[0][s] = WMMA16(a0, b[s], acc[0][s]);
      acc[1][s] = WMMA16(a1, b[s], acc[1][s]);
    }
  }

#pragma unroll
  for (int mi = 0; mi < 2; ++mi) {
#pragma unroll
    for (int s = 0; s < 4; ++s) {
#pragma unroll
      for (int g = 0; g < 8; ++g) {
        int mrow = m0 + mi * 16 + g + 8 * lh;    // 0..4095 = b*T + t
        int n    = n0 + s * 16 + lr;             // 0..3071
        f16 hv = (f16)(acc[mi][s][g] + bias[n]);
        int bb = mrow >> 10, t = mrow & (T - 1);
        if (n < C) {
          int hh = n >> 6, d = n & 63;
          qh[(((size_t)bb * H + hh) * T + t) * HD + d] = hv;
        } else if (n < 2 * C) {
          int n2 = n - C, hh = n2 >> 6, d = n2 & 63;
          kh[(((size_t)bb * H + hh) * T + t) * HD + d] = hv;
        } else {
          int n2 = n - 2 * C, hh = n2 >> 6, d = n2 & 63;
          vT[(((size_t)bb * H + hh) * HD + d) * T + t] = hv;   // transposed
        }
      }
    }
  }
}

// ---------------- flash attention with relative positions ----------------
__global__ __launch_bounds__(256) void attn_kernel(
    const f16* __restrict__ qh, const f16* __restrict__ kh,
    const f16* __restrict__ vT, const f16* __restrict__ Eh,
    f16* __restrict__ yh) {
  __shared__ float pos_lds[8][32][16];   // per-wave: rows 0..15 = P_lo, 16..31 = P_hi
  __shared__ f16   p_lds[8][16][16];

  const int lane = threadIdx.x & 31, wid = threadIdx.x >> 5;
  const int w  = blockIdx.x * 8 + wid;   // 4096 waves
  const int it = w & 63;
  const int bh = w >> 6;
  const int h  = bh & 15, bb = bh >> 4;
  const int i0 = it * 16;
  const int lr = lane & 15, lh = lane >> 4;

  const f16* qb = qh + (size_t)bh * T * HD;
  const f16* kb = kh + (size_t)bh * T * HD;
  const f16* vb = vT + (size_t)bh * HD * T;
  const f16* Eb = Eh + (size_t)h * T * HD;

  v16h qf0 = make_afrag(qb + (size_t)(i0 + lr) * HD + 0  + lh * 16);
  v16h qf1 = make_afrag(qb + (size_t)(i0 + lr) * HD + 32 + lh * 16);

  float m[8], lsum[8];
  v8f acc[4];
#pragma unroll
  for (int g = 0; g < 8; ++g) { m[g] = -1e30f; lsum[g] = 0.f; }
#pragma unroll
  for (int dg = 0; dg < 4; ++dg) acc[dg] = vzero8();

  for (int jt = 0; jt <= it; ++jt) {
    const int j0 = jt * 16;
    // --- all independent loads first: K frags, E frags, V frags ---
    v16h kf0 = make_bfrag(kb + (size_t)(j0 + lr) * HD + 0  + lh * 16);
    v16h kf1 = make_bfrag(kb + (size_t)(j0 + lr) * HD + 32 + lh * 16);

    const int rb = i0 - j0;
    int rlo = rb - 15 + lr; if (rlo < 0) rlo = 0;  // clamped rows are masked anyway
    v16h el0 = make_afrag(Eb + (size_t)rlo * HD + 0  + lh * 16);
    v16h el1 = make_afrag(Eb + (size_t)rlo * HD + 32 + lh * 16);
    const int rhi = rb + lr;                       // <= 1023 always
    v16h eh0 = make_afrag(Eb + (size_t)rhi * HD + 0  + lh * 16);
    v16h eh1 = make_afrag(Eb + (size_t)rhi * HD + 32 + lh * 16);

    int cc = j0 + lh * 16;
    if (cc + 16 > T) cc = T - 16;   // zero-multiplied K region; clamp address only
    v16h vf[4];
#pragma unroll
    for (int dg = 0; dg < 4; ++dg)
      vf[dg] = make_bfrag(vb + (size_t)(dg * 16 + lr) * T + cc);

    // --- matrix ops: S = qK^T, P_lo/P_hi = E K^T ---
    v8f S = vzero8();
    S = WMMA16(qf0, kf0, S);
    S = WMMA16(qf1, kf1, S);
    v8f Plo = vzero8();
    Plo = WMMA16(el0, kf0, Plo);
    Plo = WMMA16(el1, kf1, Plo);
    v8f Phi = vzero8();
    Phi = WMMA16(eh0, kf0, Phi);
    Phi = WMMA16(eh1, kf1, Phi);

#pragma unroll
    for (int g = 0; g < 8; ++g) {
      pos_lds[wid][g + 8 * lh][lr]      = Plo[g];  // row m -> r = rb-15+m
      pos_lds[wid][16 + g + 8 * lh][lr] = Phi[g];  // row m -> r = rb+m
    }

#pragma unroll
    for (int g = 0; g < 8; ++g) {
      int a = g + 8 * lh, bcol = lr, d = a - bcol;
      float pv = (d >= 0) ? pos_lds[wid][16 + d][bcol] : pos_lds[wid][15 + d][bcol];
      float s = (S[g] + pv) * 0.125f;              // 1/sqrt(64)
      if (jt == it && bcol > a) s = -1e30f;        // causal mask (diagonal tile only)
      S[g] = s;
    }

    // online softmax (row reductions across 16 lanes)
#pragma unroll
    for (int g = 0; g < 8; ++g) {
      float t = S[g];
      t = fmaxf(t, __shfl_xor(t, 1));
      t = fmaxf(t, __shfl_xor(t, 2));
      t = fmaxf(t, __shfl_xor(t, 4));
      t = fmaxf(t, __shfl_xor(t, 8));
      float mn = fmaxf(m[g], t);
      float alpha = __expf(m[g] - mn);
      float p = __expf(S[g] - mn);
      float ps = p;
      ps += __shfl_xor(ps, 1);
      ps += __shfl_xor(ps, 2);
      ps += __shfl_xor(ps, 4);
      ps += __shfl_xor(ps, 8);
      lsum[g] = lsum[g] * alpha + ps;
      m[g] = mn;
#pragma unroll
      for (int dg = 0; dg < 4; ++dg) acc[dg][g] *= alpha;
      p_lds[wid][g + 8 * lh][lr] = (f16)p;
    }

    // p (16x16) as A-fragment, K 16..31 zero-padded
    v8h plo = *(const v8h*)&p_lds[wid][lr][0];
    v8h phi = *(const v8h*)&p_lds[wid][lr][8];
    v16h pf;
#pragma unroll
    for (int i = 0; i < 8; ++i) { pf[i] = lh ? (f16)0 : plo[i]; pf[i + 8] = lh ? (f16)0 : phi[i]; }

#pragma unroll
    for (int dg = 0; dg < 4; ++dg) acc[dg] = WMMA16(pf, vf[dg], acc[dg]);
  }

  // write y (f16, row-major (B*T, C), head-interleaved)
#pragma unroll
  for (int dg = 0; dg < 4; ++dg) {
#pragma unroll
    for (int g = 0; g < 8; ++g) {
      int t = i0 + g + 8 * lh;
      int col = h * 64 + dg * 16 + lr;
      yh[((size_t)bb * T + t) * C + col] = (f16)(acc[dg][g] / lsum[g]);
    }
  }
}

// ---------------- projection GEMM: out = yh @ WpT^T + bp (f32) ----------------
__global__ __launch_bounds__(256) void proj_gemm(
    const f16* __restrict__ yh, const f16* __restrict__ WpT,
    const float* __restrict__ bias, float* __restrict__ out) {
  const int lane = threadIdx.x & 31, wid = threadIdx.x >> 5;
  const int w = blockIdx.x * 8 + wid;           // 2048 waves
  const int mt = w >> 4, nt = w & 15;
  const int m0 = mt * 32, n0 = nt * 64;
  const int lr = lane & 15, lh = lane >> 4;

  v8f acc[2][4];
#pragma unroll
  for (int mi = 0; mi < 2; ++mi)
#pragma unroll
    for (int s = 0; s < 4; ++s) acc[mi][s] = vzero8();

  const f16* arow0 = yh + (size_t)(m0 + lr) * C + lh * 16;
  const f16* arow1 = yh + (size_t)(m0 + 16 + lr) * C + lh * 16;
  const f16* brow[4];
#pragma unroll
  for (int s = 0; s < 4; ++s) brow[s] = WpT + (size_t)(n0 + s * 16 + lr) * C + lh * 16;

  for (int k0 = 0; k0 < C; k0 += 32) {
    __builtin_prefetch(arow0 + k0 + 32, 0, 3);
    __builtin_prefetch(brow[0] + k0 + 32, 0, 3);
    v16h a0 = make_afrag(arow0 + k0);
    v16h a1 = make_afrag(arow1 + k0);
    v16h b[4];
#pragma unroll
    for (int s = 0; s < 4; ++s) b[s] = make_bfrag(brow[s] + k0);
#pragma unroll
    for (int s = 0; s < 4; ++s) {
      acc[0][s] = WMMA16(a0, b[s], acc[0][s]);
      acc[1][s] = WMMA16(a1, b[s], acc[1][s]);
    }
  }

#pragma unroll
  for (int mi = 0; mi < 2; ++mi) {
#pragma unroll
    for (int s = 0; s < 4; ++s) {
#pragma unroll
      for (int g = 0; g < 8; ++g) {
        int mrow = m0 + mi * 16 + g + 8 * lh;
        int n = n0 + s * 16 + lr;
        out[(size_t)mrow * C + n] = acc[mi][s][g] + bias[n];
      }
    }
  }
}

extern "C" void kernel_launch(void* const* d_in, const int* in_sizes, int n_in,
                              void* d_out, int out_size, void* d_ws, size_t ws_size,
                              hipStream_t stream) {
  const float* x  = (const float*)d_in[0];  // (4,1024,1024)
  const float* Wa = (const float*)d_in[1];  // (1024,3072)
  const float* ba = (const float*)d_in[2];  // (3072,)
  const float* Wp = (const float*)d_in[3];  // (1024,1024)
  const float* bp = (const float*)d_in[4];  // (1024,)
  const float* E  = (const float*)d_in[5];  // (16,1024,64)
  float* out = (float*)d_out;

  char* wsp = (char*)d_ws;
  f16* xh  = (f16*)wsp; wsp += (size_t)4096 * 1024 * 2;
  f16* WaT = (f16*)wsp; wsp += (size_t)3072 * 1024 * 2;
  f16* WpT = (f16*)wsp; wsp += (size_t)1024 * 1024 * 2;
  f16* Eh  = (f16*)wsp; wsp += (size_t)16 * 1024 * 64 * 2;
  f16* qh  = (f16*)wsp; wsp += (size_t)64 * 1024 * 64 * 2;
  f16* kh  = (f16*)wsp; wsp += (size_t)64 * 1024 * 64 * 2;
  f16* vT  = (f16*)wsp; wsp += (size_t)64 * 64 * 1024 * 2;
  f16* yh  = (f16*)wsp; wsp += (size_t)4096 * 1024 * 2;

  cvt_f32_f16<<<(4194304 + 255) / 256, 256, 0, stream>>>(x, xh, 4194304);
  cvt_f32_f16<<<(1048576 + 255) / 256, 256, 0, stream>>>(E, Eh, 1048576);
  cvt_t_f16<<<(3145728 + 255) / 256, 256, 0, stream>>>(Wa, WaT, 1024, 3072);
  cvt_t_f16<<<(1048576 + 255) / 256, 256, 0, stream>>>(Wp, WpT, 1024, 1024);

  qkv_gemm<<<768, 256, 0, stream>>>(xh, WaT, ba, qh, kh, vT);
  attn_kernel<<<512, 256, 0, stream>>>(qh, kh, vT, Eh, yh);
  proj_gemm<<<256, 256, 0, stream>>>(yh, WpT, bp, out);
}
